// ODMixTA_layer_36867999268880
// MI455X (gfx1250) — matrix operand order repaired and checked by reading, not compiled, not run on hardware
//
#include <hip/hip_runtime.h>
#include <hip/hip_bf16.h>

// ---------------------------------------------------------------------------
// Fused MHA layer for MI455X (gfx1250, wave32, WMMA bf16), software-pipelined.
//   B=4, T=2048, D=512, H=8, hd=64
// ---------------------------------------------------------------------------

typedef __bf16 bf16_t;
typedef __bf16 v16bf __attribute__((ext_vector_type(16)));
typedef __bf16 v8bf  __attribute__((ext_vector_type(8)));
typedef float  v8f   __attribute__((ext_vector_type(8)));

#define WMMA_BF16(a, b, c)                                                    \
  __builtin_amdgcn_wmma_f32_16x16x32_bf16(false, (a), false, (b), (short)0,   \
                                          (c), false, false)

// A-matrix 16x32 bf16 fragment (blocked K layout, matches load_b_frag):
//   lanes 0-15 : K = k0+0..15, row M = lane
//   lanes 16-31: K = k0+16..31, row M = lane-16
__device__ __forceinline__ v16bf load_a_frag(const bf16_t* row, int k0, int half) {
  union { v16bf v; v8bf h[2]; } u;
  u.h[0] = *(const v8bf*)(row + k0 + half * 16);
  u.h[1] = *(const v8bf*)(row + k0 + half * 16 + 8);
  return u.v;
}

// B-matrix 32x16 bf16 fragment (blocked K layout, matches load_a_frag):
//   lanes 0-15 : K = k0+0..15, col N = lane ; lanes 16-31: K = k0+16..31
__device__ __forceinline__ v16bf load_b_frag(const bf16_t* col, int k0, int half) {
  union { v16bf v; v8bf h[2]; } u;
  u.h[0] = *(const v8bf*)(col + k0 + half * 16);
  u.h[1] = *(const v8bf*)(col + k0 + half * 16 + 8);
  return u.v;
}

__device__ __forceinline__ unsigned int pack_bf16(float a, float b) {
  union { bf16_t h[2]; unsigned int u; } x;
  x.h[0] = (bf16_t)a; x.h[1] = (bf16_t)b;
  return x.u;
}

// ---------------------------------------------------------------------------
// fp32 -> bf16 elementwise conversion
// ---------------------------------------------------------------------------
__global__ void __launch_bounds__(256) cvt_bf16_kernel(const float* __restrict__ in,
                                                       bf16_t* __restrict__ out, int n) {
  int i = blockIdx.x * 256 + threadIdx.x;
  if (i < n) out[i] = (bf16_t)in[i];
}

// ---------------------------------------------------------------------------
// out = X @ W^T + bias, written as bf16 in head-split layout.
//   transposed==0: out[(b*8+h)*2048 + t][d]      (Q, K)
//   transposed==1: out[(b*8+h)*64 + d][t]        (V^T)
// One block = 8 waves; block owns 16 M-rows, wave owns 64 N-cols.
// Double-buffered: next k-step's fragments are in flight behind the WMMAs.
// ---------------------------------------------------------------------------
__global__ void __launch_bounds__(256) qkv_gemm_kernel(
    const bf16_t* __restrict__ X, const bf16_t* __restrict__ W,
    const float* __restrict__ bias, bf16_t* __restrict__ out, int transposed) {
  const int wave = threadIdx.x >> 5;
  const int lane = threadIdx.x & 31;
  const int half = lane >> 4;
  const int l16  = lane & 15;
  const int m0 = blockIdx.x * 16;
  const int n0 = wave * 64;

  const bf16_t* arow  = X + (size_t)(m0 + l16) * 512;
  const bf16_t* brow0 = W + (size_t)(n0 +  0 + l16) * 512;
  const bf16_t* brow1 = W + (size_t)(n0 + 16 + l16) * 512;
  const bf16_t* brow2 = W + (size_t)(n0 + 32 + l16) * 512;
  const bf16_t* brow3 = W + (size_t)(n0 + 48 + l16) * 512;

  v16bf a  = load_a_frag(arow, 0, half);
  v16bf b0 = load_b_frag(brow0, 0, half);
  v16bf b1 = load_b_frag(brow1, 0, half);
  v16bf b2 = load_b_frag(brow2, 0, half);
  v16bf b3 = load_b_frag(brow3, 0, half);

  v8f acc[4] = {v8f{}, v8f{}, v8f{}, v8f{}};
#pragma unroll 2
  for (int k = 0; k < 512; k += 32) {
    const int kn = (k + 32) & 511;  // wraps to 0 on last iter (result unused)
    const v16bf an  = load_a_frag(arow, kn, half);
    const v16bf bn0 = load_b_frag(brow0, kn, half);
    const v16bf bn1 = load_b_frag(brow1, kn, half);
    const v16bf bn2 = load_b_frag(brow2, kn, half);
    const v16bf bn3 = load_b_frag(brow3, kn, half);
    acc[0] = WMMA_BF16(a, b0, acc[0]);
    acc[1] = WMMA_BF16(a, b1, acc[1]);
    acc[2] = WMMA_BF16(a, b2, acc[2]);
    acc[3] = WMMA_BF16(a, b3, acc[3]);
    a = an; b0 = bn0; b1 = bn1; b2 = bn2; b3 = bn3;
  }

#pragma unroll
  for (int t = 0; t < 4; t++) {
#pragma unroll
    for (int r = 0; r < 8; r++) {
      const int m = m0 + r + 8 * half;        // global row  (b*2048 + t_seq)
      const int n = n0 + 16 * t + l16;        // global col  (h*64 + d)
      const float val = acc[t][r] + bias[n];
      const int b  = m >> 11, tt = m & 2047;
      const int hh = n >> 6,  d  = n & 63;
      const int bh = b * 8 + hh;
      if (!transposed)
        out[((size_t)bh * 2048 + tt) * 64 + d] = (bf16_t)val;
      else
        out[((size_t)bh * 64 + d) * 2048 + tt] = (bf16_t)val;
    }
  }
}

// ---------------------------------------------------------------------------
// Flash attention, one wave per 16 query rows of one (b,h).
//   S^T = K @ Q^T, online softmax over keys, O^T = V^T @ P^T.
//   Pipelined: V loads issued before S^T WMMAs; next-chunk K loads issued
//   before the softmax VALU block; P.V WMMAs wait only on the V loads.
// ---------------------------------------------------------------------------
__global__ void __launch_bounds__(256) attn_kernel(
    const bf16_t* __restrict__ Q, const bf16_t* __restrict__ K,
    const bf16_t* __restrict__ Vt, float* __restrict__ Hout) {
  const int wave = threadIdx.x >> 5;
  const int lane = threadIdx.x & 31;
  const int half = lane >> 4;
  const int l16  = lane & 15;

  const int gid = blockIdx.x * 8 + wave;   // 4096 wave-tiles = 32 bh * 128 qt
  const int bh  = gid >> 7;
  const int q0  = (gid & 127) * 16;

  const bf16_t* Qp = Q  + (size_t)bh * 2048 * 64;
  const bf16_t* Kp = K  + (size_t)bh * 2048 * 64;
  const bf16_t* Vp = Vt + (size_t)bh * 64 * 2048;

  // Q^T as B-fragments (col = q, K-dim = d): contiguous from Q row q.
  const bf16_t* qrow = Qp + (size_t)(q0 + l16) * 64;
  const v16bf qb0 = load_b_frag(qrow, 0, half);
  const v16bf qb1 = load_b_frag(qrow, 32, half);

  // Preload K fragments for the first 32-key chunk.
  v16bf ka0 = load_a_frag(Kp + (size_t)( 0 + l16) * 64, 0, half);
  v16bf ka1 = load_a_frag(Kp + (size_t)( 0 + l16) * 64, 32, half);
  v16bf ka2 = load_a_frag(Kp + (size_t)(16 + l16) * 64, 0, half);
  v16bf ka3 = load_a_frag(Kp + (size_t)(16 + l16) * 64, 32, half);

  float mrow = -1e30f, lrow = 0.0f;
  v8f o0 = {}, o1 = {}, o2 = {}, o3 = {};
  const float scale = 0.125f;  // 1/sqrt(64)

  for (int kc = 0; kc < 2048; kc += 32) {
    // V^T fragments for this chunk, issued early (consumed after softmax).
    const v16bf va0 = load_a_frag(Vp + (size_t)( 0 + l16) * 2048, kc, half);
    const v16bf va1 = load_a_frag(Vp + (size_t)(16 + l16) * 2048, kc, half);
    const v16bf va2 = load_a_frag(Vp + (size_t)(32 + l16) * 2048, kc, half);
    const v16bf va3 = load_a_frag(Vp + (size_t)(48 + l16) * 2048, kc, half);

    // S^T tiles from K fragments preloaded last iteration.
    v8f st0 = {}, st1 = {};
    st0 = WMMA_BF16(ka0, qb0, st0);
    st0 = WMMA_BF16(ka1, qb1, st0);
    st1 = WMMA_BF16(ka2, qb0, st1);
    st1 = WMMA_BF16(ka3, qb1, st1);

    // Prefetch next chunk's K fragments (wrapped index; extra load harmless).
    const int kn = (kc + 32) & 2047;
    const bf16_t* krn0 = Kp + (size_t)(kn + l16) * 64;
    const bf16_t* krn1 = Kp + (size_t)(kn + 16 + l16) * 64;
    ka0 = load_a_frag(krn0, 0, half);
    ka1 = load_a_frag(krn0, 32, half);
    ka2 = load_a_frag(krn1, 0, half);
    ka3 = load_a_frag(krn1, 32, half);

    // Per-q chunk max (keys split across half-waves -> one xor16 shuffle).
    float cm = -1e30f;
#pragma unroll
    for (int r = 0; r < 8; r++) {
      st0[r] *= scale; st1[r] *= scale;
      cm = fmaxf(cm, fmaxf(st0[r], st1[r]));
    }
    cm = fmaxf(cm, __shfl_xor(cm, 16, 32));
    const float mnew  = fmaxf(mrow, cm);
    const float alpha = __expf(mrow - mnew);
    mrow = mnew;

    float p0[8], p1[8], ls = 0.0f;
#pragma unroll
    for (int r = 0; r < 8; r++) {
      p0[r] = __expf(st0[r] - mnew);
      p1[r] = __expf(st1[r] - mnew);
      ls += p0[r] + p1[r];
    }
    ls += __shfl_xor(ls, 16, 32);
    lrow = lrow * alpha + ls;
#pragma unroll
    for (int r = 0; r < 8; r++) {
      o0[r] *= alpha; o1[r] *= alpha; o2[r] *= alpha; o3[r] *= alpha;
    }

    // Build P^T B-fragment (col = q, K-dim = key). C-layout already has the
    // right q-per-lane; only the 8 "other-half" keys come via shfl_xor(16).
    unsigned int pk0[4], pk1[4];
#pragma unroll
    for (int j = 0; j < 4; j++) {
      pk0[j] = pack_bf16(p0[2 * j], p0[2 * j + 1]);
      pk1[j] = pack_bf16(p1[2 * j], p1[2 * j + 1]);
    }
    union { v16bf v; unsigned int u[8]; } pb;
#pragma unroll
    for (int j = 0; j < 4; j++) {
      const unsigned int x0 = __shfl_xor(pk0[j], 16, 32);
      const unsigned int x1 = __shfl_xor(pk1[j], 16, 32);
      pb.u[j]     = half ? x1     : pk0[j];  // keys 16h + {0..7}
      pb.u[4 + j] = half ? pk1[j] : x0;      // keys 16h + {8..15}
    }

    // O^T += V^T(tile) @ P^T  (waits only on this chunk's V loads).
    o0 = WMMA_BF16(va0, pb.v, o0);
    o1 = WMMA_BF16(va1, pb.v, o1);
    o2 = WMMA_BF16(va2, pb.v, o2);
    o3 = WMMA_BF16(va3, pb.v, o3);
  }

  const float linv = 1.0f / lrow;
  const int b = bh >> 3, hh = bh & 7;
  float* orow = Hout + (size_t)(b * 2048 + q0 + l16) * 512 + hh * 64;
#pragma unroll
  for (int r = 0; r < 8; r++) {
    orow[ 0 + 8 * half + r] = o0[r] * linv;
    orow[16 + 8 * half + r] = o1[r] * linv;
    orow[32 + 8 * half + r] = o2[r] * linv;
    orow[48 + 8 * half + r] = o3[r] * linv;
  }
}

// ---------------------------------------------------------------------------
// out = LayerNorm(H + X) * gamma + beta ; one wave per 512-wide row.
// ---------------------------------------------------------------------------
__global__ void __launch_bounds__(256) ln_kernel(
    const float* __restrict__ H, const float* __restrict__ X,
    const float* __restrict__ gamma, const float* __restrict__ beta,
    float* __restrict__ out) {
  const int wave = threadIdx.x >> 5;
  const int lane = threadIdx.x & 31;
  const int row  = blockIdx.x * 8 + wave;
  const float* h = H + (size_t)row * 512;
  const float* x = X + (size_t)row * 512;

  float vals[16], s = 0.0f, s2 = 0.0f;
#pragma unroll
  for (int i = 0; i < 16; i++) {
    const int idx = i * 32 + lane;
    const float v = h[idx] + x[idx];
    vals[i] = v; s += v; s2 += v * v;
  }
#pragma unroll
  for (int off = 16; off; off >>= 1) {
    s  += __shfl_xor(s, off, 32);
    s2 += __shfl_xor(s2, off, 32);
  }
  const float mu   = s * (1.0f / 512.0f);
  const float var  = s2 * (1.0f / 512.0f) - mu * mu;
  const float rstd = rsqrtf(var + 1e-5f);
  float* o = out + (size_t)row * 512;
#pragma unroll
  for (int i = 0; i < 16; i++) {
    const int idx = i * 32 + lane;
    o[idx] = (vals[i] - mu) * rstd * gamma[idx] + beta[idx];
  }
}

// ---------------------------------------------------------------------------
extern "C" void kernel_launch(void* const* d_in, const int* in_sizes, int n_in,
                              void* d_out, int out_size, void* d_ws, size_t ws_size,
                              hipStream_t stream) {
  (void)in_sizes; (void)n_in; (void)out_size; (void)ws_size;
  const float* x     = (const float*)d_in[0];
  const float* Wq    = (const float*)d_in[1];
  const float* bq    = (const float*)d_in[2];
  const float* Wk    = (const float*)d_in[3];
  const float* bk    = (const float*)d_in[4];
  const float* Wv    = (const float*)d_in[5];
  const float* bv    = (const float*)d_in[6];
  const float* gamma = (const float*)d_in[7];
  const float* beta  = (const float*)d_in[8];
  float* out = (float*)d_out;

  char* ws = (char*)d_ws;
  // Workspace layout (bytes), total ~52 MB:
  bf16_t* XB  = (bf16_t*)(ws + 0);          //  8,388,608  x  bf16 [8192,512]
  bf16_t* WQB = (bf16_t*)(ws +  8388608);   //    524,288
  bf16_t* WKB = (bf16_t*)(ws +  8912896);   //    524,288
  bf16_t* WVB = (bf16_t*)(ws +  9437184);   //    524,288
  bf16_t* QB  = (bf16_t*)(ws +  9961472);   //  8,388,608  [32][2048][64]
  bf16_t* KB  = (bf16_t*)(ws + 18350080);   //  8,388,608  [32][2048][64]
  bf16_t* VTB = (bf16_t*)(ws + 26738688);   //  8,388,608  [32][64][2048]
  float*  HB  = (float*)(ws + 35127296);    // 16,777,216  [8192][512]

  const int NX = 4 * 2048 * 512;  // 4,194,304
  const int NW = 512 * 512;       //   262,144

  cvt_bf16_kernel<<<NX / 256, 256, 0, stream>>>(x, XB, NX);
  cvt_bf16_kernel<<<NW / 256, 256, 0, stream>>>(Wq, WQB, NW);
  cvt_bf16_kernel<<<NW / 256, 256, 0, stream>>>(Wk, WKB, NW);
  cvt_bf16_kernel<<<NW / 256, 256, 0, stream>>>(Wv, WVB, NW);

  qkv_gemm_kernel<<<512, 256, 0, stream>>>(XB, WQB, bq, QB, 0);
  qkv_gemm_kernel<<<512, 256, 0, stream>>>(XB, WKB, bk, KB, 0);
  qkv_gemm_kernel<<<512, 256, 0, stream>>>(XB, WVB, bv, VTB, 1);

  attn_kernel<<<512, 256, 0, stream>>>(QB, KB, VTB, HB);

  ln_kernel<<<1024, 256, 0, stream>>>(HB, x, gamma, beta, out);
}
